// EdgeModel_70600672411926
// MI455X (gfx1250) — matrix operand order repaired and checked by reading, not compiled, not run on hardware
//
#include <hip/hip_runtime.h>

typedef __attribute__((ext_vector_type(16))) _Float16 v16h;
typedef __attribute__((ext_vector_type(8)))  float    v8f;

#define EDIM      32
#define TILE_ROWS 64
#define XLD       132   // padded f32 row stride for x tile (bank-conflict free)
#define HLD       136   // padded f16 row stride for h tile

// Convert w1 [k][n] and w2 [k][n] (f32, 128x128) to transposed f16 wT[n][k].
__global__ void prep_weights(const float* __restrict__ w1,
                             const float* __restrict__ w2,
                             _Float16* __restrict__ wT) {
  int idx = blockIdx.x * blockDim.x + threadIdx.x;
  if (idx < 128 * 128) {
    int n = idx >> 7, k = idx & 127;
    wT[idx]         = (_Float16)w1[k * 128 + n];   // w1T
    wT[16384 + idx] = (_Float16)w2[k * 128 + n];   // w2T
  }
}

__launch_bounds__(128)
__global__ void edge_mlp_kernel(const float* __restrict__ src,
                                const float* __restrict__ dstn,
                                const float* __restrict__ ea,
                                const float* __restrict__ u,
                                const int*   __restrict__ batch,
                                const float* __restrict__ b1,
                                const float* __restrict__ b2,
                                const _Float16* __restrict__ wT,  // [0..16384) w1T, [16384..32768) w2T
                                float* __restrict__ out,
                                int E)
{
  __shared__ float    x_lds[TILE_ROWS * XLD];
  __shared__ _Float16 h_lds[TILE_ROWS * HLD];

  const int t  = threadIdx.x;
  const int e0 = blockIdx.x * TILE_ROWS;

  // ---- cooperative gather + concat of the x tile (f32) into LDS ----
  // 64 rows x 32 float4-groups = 2048 float4; 128 threads -> 16 each
#pragma unroll
  for (int i = 0; i < 16; ++i) {
    int idx = i * 128 + t;
    int row = idx >> 5;          // / 32 groups per row
    int g   = idx & 31;          // float4 group within row
    int seg = g >> 3;            // which 32-wide source segment
    int off = (g & 7) * 4;       // float offset within segment
    int e   = e0 + row;
    float4 v = make_float4(0.f, 0.f, 0.f, 0.f);
    if (e < E) {
      const float* p;
      if      (seg == 0) p = src  + (size_t)e * EDIM + off;
      else if (seg == 1) p = dstn + (size_t)e * EDIM + off;
      else if (seg == 2) p = ea   + (size_t)e * EDIM + off;
      else               p = u    + (size_t)batch[e] * EDIM + off;
      v = *(const float4*)p;
    }
    *(float4*)&x_lds[row * XLD + g * 4] = v;   // 16B-aligned (XLD = 4*33)
  }
  __syncthreads();

  const int wave = t >> 5;
  const int lane = t & 31;
  const int lrow = lane & 15;   // A: row m / B: col n / C: col n
  const int lhi  = lane >> 4;   // half-wave selector per ISA layouts
  const int m0   = wave * 16;

  v8f acc[8];

  // ---- GEMM1: h = relu(x @ w1 + b1) ----
#pragma unroll
  for (int nf = 0; nf < 8; ++nf) {
    float bv = b1[nf * 16 + lrow];
#pragma unroll
    for (int r = 0; r < 8; ++r) acc[nf][r] = bv;
  }
#pragma unroll
  for (int kt = 0; kt < 4; ++kt) {
    // A fragment: lane holds m = lrow, 16 contiguous K at kt*32 + lhi*16
    const float* xr = &x_lds[(m0 + lrow) * XLD + kt * 32 + lhi * 16];
    v16h a;
#pragma unroll
    for (int j = 0; j < 8; ++j) { a[j] = (_Float16)xr[j]; a[8 + j] = (_Float16)xr[8 + j]; }
#pragma unroll
    for (int nf = 0; nf < 8; ++nf) {
      // B fragment: lane holds col n = lrow, 16 contiguous K at kt*32 + lhi*16
      const v16h b = *(const v16h*)(wT + ((size_t)(nf * 16 + lrow) * 128 + kt * 32 + lhi * 16));
      acc[nf] = __builtin_amdgcn_wmma_f32_16x16x32_f16(
          false, a, false, b, (short)0, acc[nf], false, false);
    }
  }
  // ReLU -> h tile (f16) in LDS; C layout: row = r + lhi*8, col = lrow
#pragma unroll
  for (int nf = 0; nf < 8; ++nf)
#pragma unroll
    for (int r = 0; r < 8; ++r)
      h_lds[(m0 + r + lhi * 8) * HLD + nf * 16 + lrow] = (_Float16)fmaxf(acc[nf][r], 0.f);
  __syncthreads();

  // ---- GEMM2: out = h @ w2 + b2 + x ----
  const _Float16* w2T = wT + 16384;
#pragma unroll
  for (int nf = 0; nf < 8; ++nf) {
    float bv = b2[nf * 16 + lrow];
#pragma unroll
    for (int r = 0; r < 8; ++r) acc[nf][r] = bv;
  }
#pragma unroll
  for (int kt = 0; kt < 4; ++kt) {
    const _Float16* hr = &h_lds[(m0 + lrow) * HLD + kt * 32 + lhi * 16];
    v16h a;
#pragma unroll
    for (int j = 0; j < 8; ++j) { a[j] = hr[j]; a[8 + j] = hr[8 + j]; }
#pragma unroll
    for (int nf = 0; nf < 8; ++nf) {
      const v16h b = *(const v16h*)(w2T + ((size_t)(nf * 16 + lrow) * 128 + kt * 32 + lhi * 16));
      acc[nf] = __builtin_amdgcn_wmma_f32_16x16x32_f16(
          false, a, false, b, (short)0, acc[nf], false, false);
    }
  }
  // residual (exact f32 x from LDS) + store
#pragma unroll
  for (int nf = 0; nf < 8; ++nf) {
#pragma unroll
    for (int r = 0; r < 8; ++r) {
      int m = m0 + r + lhi * 8;
      int e = e0 + m;
      if (e < E) {
        out[(size_t)e * 128 + nf * 16 + lrow] =
            acc[nf][r] + x_lds[m * XLD + nf * 16 + lrow];
      }
    }
  }
}

extern "C" void kernel_launch(void* const* d_in, const int* in_sizes, int n_in,
                              void* d_out, int out_size, void* d_ws, size_t ws_size,
                              hipStream_t stream) {
  const float* src = (const float*)d_in[0];
  const float* dst = (const float*)d_in[1];
  const float* ea  = (const float*)d_in[2];
  const float* u   = (const float*)d_in[3];
  const int*   bat = (const int*)  d_in[4];
  const float* w1  = (const float*)d_in[5];
  const float* b1  = (const float*)d_in[6];
  const float* w2  = (const float*)d_in[7];
  const float* b2  = (const float*)d_in[8];
  float* out = (float*)d_out;

  int E = in_sizes[0] / EDIM;

  _Float16* wT = (_Float16*)d_ws;   // 64 KB scratch: w1T f16 then w2T f16

  prep_weights<<<64, 256, 0, stream>>>(w1, w2, wT);

  int blocks = (E + TILE_ROWS - 1) / TILE_ROWS;
  edge_mlp_kernel<<<blocks, 128, 0, stream>>>(src, dst, ea, u, bat, b1, b2, wT, out, E);
}
